// sLSTMCell_31044023616494
// MI455X (gfx1250) — matrix operand order repaired and checked by reading, not compiled, not run on hardware
//
#include <hip/hip_runtime.h>
#include <math.h>

// ---------------------------------------------------------------------------
// sLSTM (xLSTM) recurrence for MI455X / gfx1250.
//   T=512, B=16, H=4 heads, D=256, HID=1024, G=4 gates.
// Persistent kernel: 16 workgroups, each owns 64 hidden units (x all 4 gates)
// = a 16x256 slice of the per-step 16x4096 recurrent GEMM.
// GEMM runs on v_wmma_f32_16x16x32_bf16 with split-bf16 (hi/lo) operands:
//   h*R ~= h_hi*R_hi + h_hi*R_lo + h_lo*R_hi   (fp32 accumulate)
// Weights are pre-packed into WMMA B-fragment layout in d_ws (L2-resident).
// Cross-step sync: atomic grid barrier + double-buffered global h.
// ---------------------------------------------------------------------------

typedef __attribute__((ext_vector_type(16))) __bf16 v16bf;
typedef __attribute__((ext_vector_type(8)))  float  v8f;

#define TSTEPS   512
#define BATCH    16
#define HIDDEN   1024
#define NCOL     4096         // G * HIDDEN
#define NWG      16           // workgroups (unit blocks of 64 units)
#define NTHREADS 256          // 8 waves of 32
#define NPACK    (16*16*8*32*16)  // wg * tile * kchunk * lane * 16 bf16 = 1048576

// d_ws byte layout:
//   [0,      131072)  : h double buffer, 2 * 16 * 1024 f32
//   [131072, 131076)  : grid-barrier counter (u32)
//   [132096, +2MB)    : packed W_hi  (bf16, WMMA B-fragment layout)
//   [+2MB,   +4MB)    : packed W_lo
#define WS_HBUF_OFF 0
#define WS_CNT_OFF  131072
#define WS_WHI_OFF  132096

// ---------------------------------------------------------------------------
// Prep: rearrange recurrent_kernel (H,D,G,D) f32 into per-fragment bf16 hi/lo
// matching the reference's rk2[h, k, g*256+dout] semantics:
//   rk2[h,k,g*256+dout] = rk[h, 4*(k&63)+g, dout>>6, 4*(dout&63)+(k>>6)]
// Fragment layout (B-matrix 32x16 bf16, one 32-byte run per lane):
//   lane<16 : col N=lane,    K = kc*32 + (0..15)
//   lane>=16: col N=lane-16, K = kc*32 + (16..31)
// Also zeroes the h ping-pong buffers and the barrier counter (required every
// launch: the harness does not re-initialize d_ws between replays).
// ---------------------------------------------------------------------------
__global__ void slstm_prep(const float* __restrict__ rk,
                           float* __restrict__ hbuf,
                           unsigned* __restrict__ cnt,
                           __bf16* __restrict__ whi,
                           __bf16* __restrict__ wlo)
{
    const size_t id = (size_t)blockIdx.x * blockDim.x + threadIdx.x;
    if (id < 2 * BATCH * HIDDEN) hbuf[id] = 0.0f;   // both h buffers
    if (id == 0) *cnt = 0u;
    if (id < (size_t)NPACK) {
        const int e    = (int)(id & 15);
        const int l    = (int)((id >> 4) & 31);
        const int kc   = (int)((id >> 9) & 7);
        const int tile = (int)((id >> 12) & 15);
        const int wg   = (int)(id >> 16);
        const int N    = (l < 16) ? l : (l - 16);
        const int k    = kc * 32 + ((l < 16) ? e : (16 + e));
        const int c    = tile * 16 + N;     // local column 0..255
        const int g    = c >> 6;            // gate
        const int ul   = c & 63;            // unit within block
        const int unit = wg * 64 + ul;      // global unit 0..1023
        const int head = wg >> 2;
        const int dout = unit & 255;        // unit within head
        const int d1   = 4 * (k & 63) + g;
        const int g1   = dout >> 6;
        const int d2   = 4 * (dout & 63) + (k >> 6);
        const float v  = rk[(((size_t)head * 256 + d1) * 4 + g1) * 256 + d2];
        const __bf16 hi = (__bf16)v;
        whi[id] = hi;
        wlo[id] = (__bf16)(v - (float)hi);
    }
}

// ---------------------------------------------------------------------------
// Main persistent recurrence kernel.
// ---------------------------------------------------------------------------
__global__ __launch_bounds__(NTHREADS) void slstm_main(
    const float* __restrict__ x,        // (T, B, 4096)
    const float* __restrict__ bias,     // (H, G, D) flat
    float* __restrict__ out,            // outs (T,B,1024) then state (4,B,1024)
    float* __restrict__ hbuf,           // 2 x (B,1024) ping-pong
    unsigned* __restrict__ cnt,
    const __bf16* __restrict__ whi,
    const __bf16* __restrict__ wlo)
{
    // A-operand fragments (16x32 bf16, contiguous k: lane half h holds K = 16h..16h+15 of each chunk) + gate staging.
    __shared__ __align__(32) __bf16 Ahi[8][32][16];   // [kchunk][lane][16 bf16]
    __shared__ __align__(32) __bf16 Alo[8][32][16];
    __shared__ float rawS[4][64][16];                 // [gate][unit][batch]

    const int tid  = threadIdx.x;
    const int wg   = blockIdx.x;        // unit block 0..15
    const int head = wg >> 2;
    const int wave = tid >> 5;
    const int lane = tid & 31;

    // Gate-phase ownership: one unit, 4 batch rows per thread.
    const int ul   = tid & 63;
    const int unit = wg * 64 + ul;
    const int dout = unit & 255;
    const int b0   = tid >> 6;

    float h_s[4], c_s[4], n_s[4], m_s[4];
    #pragma unroll
    for (int j = 0; j < 4; ++j) { h_s[j] = 0.f; c_s[j] = 0.f; n_s[j] = 0.f; m_s[j] = 0.f; }

    float bz[4];
    #pragma unroll
    for (int g = 0; g < 4; ++g) bz[g] = bias[head * 1024 + g * 256 + dout];

    // A-fill mapping: 256 threads <-> (kchunk 0..7) x (lane 0..31).
    const int a_kc = tid >> 5;
    const int a_l  = tid & 31;
    const int a_b  = a_l & 15;          // M row = batch

    for (int t = 0; t < TSTEPS; ++t) {
        const int p = t & 1;
        const float* __restrict__ hread  = hbuf + p * (BATCH * HIDDEN);
        float* __restrict__       hwrite = hbuf + (1 - p) * (BATCH * HIDDEN);

        // ---- Phase A: stage h (this head's 16x256 slice) as bf16 hi/lo A-frags.
        #pragma unroll
        for (int e = 0; e < 16; ++e) {
            // lane<16: K = 0..15 ; lane>=16: K = 16..31 (matches B packing)
            int k;
            if (a_l < 16) k = a_kc * 32 + e;
            else          k = a_kc * 32 + 16 + e;
            const float v = hread[a_b * 1024 + head * 256 + k];
            const __bf16 hi = (__bf16)v;
            Ahi[a_kc][a_l][e] = hi;
            Alo[a_kc][a_l][e] = (__bf16)(v - (float)hi);
        }
        __syncthreads();

        // ---- Phase B: 2 output tiles per wave, K=256 in 8 chunks, 3 products.
        #pragma unroll
        for (int tt = 0; tt < 2; ++tt) {
            const int tile = wave * 2 + tt;
            v8f acc = {0.f, 0.f, 0.f, 0.f, 0.f, 0.f, 0.f, 0.f};
            const size_t wbase = ((size_t)(wg * 16 + tile)) * (8 * 32 * 16);
            #pragma unroll
            for (int kc = 0; kc < 8; ++kc) {
                const v16bf ah = *reinterpret_cast<const v16bf*>(&Ahi[kc][lane][0]);
                const v16bf al = *reinterpret_cast<const v16bf*>(&Alo[kc][lane][0]);
                const size_t foff = wbase + ((size_t)kc * 32 + lane) * 16;
                const v16bf bh = *reinterpret_cast<const v16bf*>(whi + foff);
                const v16bf bl = *reinterpret_cast<const v16bf*>(wlo + foff);
                acc = __builtin_amdgcn_wmma_f32_16x16x32_bf16(false, al, false, bh,
                                                              (short)0, acc, false, false);
                acc = __builtin_amdgcn_wmma_f32_16x16x32_bf16(false, ah, false, bl,
                                                              (short)0, acc, false, false);
                acc = __builtin_amdgcn_wmma_f32_16x16x32_bf16(false, ah, false, bh,
                                                              (short)0, acc, false, false);
            }
            // Scatter C/D (8 VGPRs: lane 0-15 rows M=r, lane 16-31 rows M=r+8).
            const int N  = lane & 15;
            const int c  = tile * 16 + N;
            const int gg = c >> 6;
            const int uu = c & 63;
            const int bb = (lane >> 4) * 8;
            #pragma unroll
            for (int r = 0; r < 8; ++r)
                rawS[gg][uu][bb + r] = acc[r];
        }
        __syncthreads();

        // ---- Phase C: gates + state update for (unit, 4 batches).
        #pragma unroll
        for (int j = 0; j < 4; ++j) {
            const int b = j * 4 + b0;
            const size_t xrow = ((size_t)t * BATCH + b) * (size_t)NCOL;
            float pre[4];
            #pragma unroll
            for (int g = 0; g < 4; ++g)
                pre[g] = rawS[g][ul][b] + x[xrow + (size_t)g * HIDDEN + unit] + bz[g];
            if (t + 1 < TSTEPS) {   // hide next step's Wx latency
                const size_t xrow2 = ((size_t)(t + 1) * BATCH + b) * (size_t)NCOL;
                #pragma unroll
                for (int g = 0; g < 4; ++g)
                    __builtin_prefetch(&x[xrow2 + (size_t)g * HIDDEN + unit], 0, 1);
            }
            const float iraw = pre[0], fraw = pre[1], zraw = pre[2], oraw = pre[3];
            const float lsf  = fminf(fraw, 0.f) - log1pf(expf(-fabsf(fraw)));
            const float logfplusm = m_s[j] + lsf;
            // all(n==0) is true only at t==0 (n becomes >0 after the first step)
            const float mnew  = (t == 0) ? iraw : fmaxf(iraw, logfplusm);
            const float ogate = 1.f / (1.f + expf(-oraw));
            const float igate = expf(iraw - mnew);
            const float fgate = expf(logfplusm - mnew);
            const float cnew  = fgate * c_s[j] + igate * tanhf(zraw);
            const float nnew  = fgate * h_s[j] + igate;
            const float hnew  = ogate * cnew / nnew;
            h_s[j] = hnew; c_s[j] = cnew; n_s[j] = nnew; m_s[j] = mnew;
            hwrite[b * HIDDEN + unit] = hnew;
            out[((size_t)t * BATCH + b) * HIDDEN + unit] = hnew;
        }

        // ---- Grid barrier (monotonic counter; 16 co-resident blocks).
        __threadfence();
        __syncthreads();
        if (tid == 0) {
            atomicAdd(cnt, 1u);
            const unsigned target = (unsigned)(NWG * (t + 1));
            while (atomicAdd(cnt, 0u) < target) { __builtin_amdgcn_s_sleep(1); }
        }
        __syncthreads();
        __threadfence();
    }

    // ---- Final state: stack (h, c, n, m) after outs.
    const size_t base = (size_t)TSTEPS * BATCH * HIDDEN;
    #pragma unroll
    for (int j = 0; j < 4; ++j) {
        const int b = j * 4 + b0;
        out[base + 0 * BATCH * HIDDEN + b * HIDDEN + unit] = h_s[j];
        out[base + 1 * BATCH * HIDDEN + b * HIDDEN + unit] = c_s[j];
        out[base + 2 * BATCH * HIDDEN + b * HIDDEN + unit] = n_s[j];
        out[base + 3 * BATCH * HIDDEN + b * HIDDEN + unit] = m_s[j];
    }
}

// ---------------------------------------------------------------------------
extern "C" void kernel_launch(void* const* d_in, const int* in_sizes, int n_in,
                              void* d_out, int out_size, void* d_ws, size_t ws_size,
                              hipStream_t stream)
{
    const float* x    = (const float*)d_in[0];   // (512,16,4096) f32
    const float* rk   = (const float*)d_in[1];   // (4,256,4,256) f32
    const float* bias = (const float*)d_in[2];   // (4,4,256)     f32
    float* out = (float*)d_out;

    char* ws = (char*)d_ws;
    float*    hbuf = (float*)(ws + WS_HBUF_OFF);
    unsigned* cnt  = (unsigned*)(ws + WS_CNT_OFF);
    __bf16*   whi  = (__bf16*)(ws + WS_WHI_OFF);
    __bf16*   wlo  = whi + NPACK;

    // Pack weights + zero h buffers + reset barrier counter (every call).
    slstm_prep<<<NPACK / 256, 256, 0, stream>>>(rk, hbuf, cnt, whi, wlo);
    // Persistent recurrence: 16 WGPs, 8 waves each.
    slstm_main<<<NWG, NTHREADS, 0, stream>>>(x, bias, out, hbuf, cnt, whi, wlo);
}
